// QuantumKernelMethod_65481071397762
// MI455X (gfx1250) — matrix operand order repaired and checked by reading, not compiled, not run on hardware
//
#include <hip/hip_runtime.h>

// ---------------------------------------------------------------------------
// RBF kernel K[i,j] = exp(-gamma * ||x_i - y_j||^2), gamma = 1.0
// x: [8192,512] f32, y: [8192,512] f32, out: [8192,8192] f32
//
// Pipeline:
//   1) split_bf16_kernel : X,Y f32 -> (hi,lo) bf16 arrays in d_ws (one pass,
//      bandwidth bound; arrays fit in MI455X's 192MB L2 afterwards)
//   2) row_sqnorm_kernel : per-row ||.||^2 into d_ws
//   3) rbf_wmma_kernel   : S = X*Y^T via bf16x3 split WMMA (hh + hl + lh),
//      tiles staged LDS-side by the Tensor Data Mover (tensor_load_to_lds)
//      with TDM padding producing a 40-bf16 row stride (bank-conflict free),
//      double buffered, fused exp epilogue with non-temporal stores.
// ---------------------------------------------------------------------------

typedef __attribute__((ext_vector_type(16))) __bf16    v16bf;
typedef __attribute__((ext_vector_type(8)))  __bf16    v8bf;
typedef __attribute__((ext_vector_type(8)))  float     v8f;
typedef __attribute__((ext_vector_type(4)))  unsigned  v4u;
typedef __attribute__((ext_vector_type(8)))  unsigned  v8u;

#define K_DIM   512
#define N_ROWS  8192
#define TILE    128      // output tile per workgroup (128x128)
#define KTILE   32       // K step per stage (matches wmma 16x16x32)
#define LDSK    40       // padded LDS row stride in bf16 elems (80B = 20 banks)
#define TILE_B  (TILE * LDSK * 2)        // bytes per staged tile (10240)
#define BUF_B   (4 * TILE_B)             // bytes per LDS buffer (40960)
#define GAMMA   1.0f

union Frag { v16bf v; v8bf h[2]; };

__device__ __forceinline__ __bf16 f2bf(float f) {
  unsigned u = __builtin_bit_cast(unsigned, f);
  u += 0x7FFFu + ((u >> 16) & 1u);            // round-to-nearest-even
  unsigned short s = (unsigned short)(u >> 16);
  return __builtin_bit_cast(__bf16, s);
}
__device__ __forceinline__ float bf2f(__bf16 h) {
  unsigned short s = __builtin_bit_cast(unsigned short, h);
  unsigned u = ((unsigned)s) << 16;
  return __builtin_bit_cast(float, u);
}

// ---------------------------------------------------------------------------
// One-shot f32 -> bf16 (hi, lo) split. 8 elements / thread.
// ---------------------------------------------------------------------------
__global__ __launch_bounds__(256) void split_bf16_kernel(
    const float* __restrict__ src, __bf16* __restrict__ hi,
    __bf16* __restrict__ lo) {
  const size_t base = ((size_t)blockIdx.x * 256 + threadIdx.x) * 8;
  const float4* s4 = (const float4*)(src + base);
  float4 a = s4[0], b = s4[1];
  float f[8] = {a.x, a.y, a.z, a.w, b.x, b.y, b.z, b.w};
  v8bf h, l;
  #pragma unroll
  for (int i = 0; i < 8; ++i) {
    __bf16 hv = f2bf(f[i]);
    h[i] = hv;
    l[i] = f2bf(f[i] - bf2f(hv));
  }
  *(v8bf*)(hi + base) = h;
  *(v8bf*)(lo + base) = l;
}

// ---------------------------------------------------------------------------
// Row squared-norm: one wave32 per row of a [rows, 512] f32 matrix.
// ---------------------------------------------------------------------------
__global__ __launch_bounds__(256) void row_sqnorm_kernel(
    const float* __restrict__ v, float* __restrict__ out) {
  const int lane = threadIdx.x & 31;
  const int row  = blockIdx.x * 8 + (threadIdx.x >> 5);
  const float4* p = (const float4*)(v + (size_t)row * K_DIM);
  float s = 0.0f;
  #pragma unroll
  for (int c = 0; c < 4; ++c) {
    float4 f = p[lane + 32 * c];
    s += f.x * f.x + f.y * f.y + f.z * f.z + f.w * f.w;
  }
  #pragma unroll
  for (int off = 16; off > 0; off >>= 1) s += __shfl_xor(s, off, 32);
  if (lane == 0) out[row] = s;
}

// ---------------------------------------------------------------------------
// Issue one TDM 2D tile load: 128 rows x 32 bf16 from a [*,512] bf16 array
// into LDS at lds_byte_off. TDM padding: after every 16 DWORDs (one 32-elem
// row, 64B) insert 4 DWORDs (16B) -> LDS row stride = 40 bf16 = 80B.
// Descriptor bitfields per CDNA5 ISA.
// The LDS base pointer is passed as an (unused) asm operand so the shared
// buffer escapes: with the "memory" clobber the compiler must assume the
// TDM writes it, keeping the subsequent ds_load fragment reads alive.
// ---------------------------------------------------------------------------
__device__ __forceinline__ void tdm_load_tile(const __bf16* lds_marker,
                                              unsigned lds_byte_off,
                                              const __bf16* gtile) {
  const unsigned long long ga = (unsigned long long)(size_t)gtile;
  v4u g0;
  g0[0] = 1u;                                    // count=1 (valid), user mode
  g0[1] = lds_byte_off;                          // lds_addr (bytes)
  g0[2] = (unsigned)(ga & 0xFFFFFFFFu);          // global_addr[31:0]
  g0[3] = (unsigned)((ga >> 32) & 0x01FFFFFFu)   // global_addr[56:32]
        | (2u << 30);                            // type=2 ("image")
  v8u g1;
  g1[0] = (1u << 16)                             // data_size = 1 (2 bytes)
        | (1u << 20)                             // pad_enable
        | (3u << 22)                             // pad_interval: 16 DWORDs
        | (3u << 25);                            // pad_amount : 4 DWORDs
  g1[1] = ((unsigned)K_DIM & 0xFFFFu) << 16;     // tensor_dim0[15:0] = 512
  g1[2] = ((unsigned)N_ROWS & 0xFFFFu) << 16;    // dim0 hi=0 | tensor_dim1 lo
  g1[3] = ((unsigned)N_ROWS >> 16)               // tensor_dim1[31:16]
        | ((unsigned)KTILE << 16);               // tile_dim0 = 32
  g1[4] = (unsigned)TILE;                        // tile_dim1 = 128, tile_dim2=0
  g1[5] = (unsigned)K_DIM;                       // tensor_dim0_stride = 512
  g1[6] = 0u;                                    // stride hi / dim1_stride lo
  g1[7] = 0u;                                    // dim1_stride hi (unused, 2D)
  asm volatile("tensor_load_to_lds %0, %1"
               :
               : "s"(g0), "s"(g1), "s"(lds_marker)
               : "memory");
}

// ---------------------------------------------------------------------------
// Main fused kernel: 128x128 tile per workgroup, 8 waves in 4(M) x 2(N),
// each wave computes 2x4 16x16 wmma tiles with bf16x3 split precision.
// TDM double-buffered staging: wave 0 drives the DMA.
// ---------------------------------------------------------------------------
__global__ __launch_bounds__(256) void rbf_wmma_kernel(
    const __bf16* __restrict__ xh, const __bf16* __restrict__ xl,
    const __bf16* __restrict__ yh, const __bf16* __restrict__ yl,
    const float* __restrict__ xsq, const float* __restrict__ ysq,
    float* __restrict__ out, int ldo) {
  __shared__ __bf16 smem[2 * BUF_B / 2];        // 80 KB, two buffers

  const int t       = threadIdx.x;
  const int lane    = t & 31;
  const int wave    = t >> 5;          // 0..7
  const int wm      = wave >> 1;       // 0..3 (M direction)
  const int wn      = wave & 1;        // 0..1 (N direction)
  const int laneRow = lane & 15;
  const int half    = lane >> 4;       // 0 or 1

  const int bm = blockIdx.x * TILE;    // rows of x
  const int bn = blockIdx.y * TILE;    // rows of y (columns of out)

  const size_t xoff = (size_t)bm * K_DIM;
  const size_t yoff = (size_t)bn * K_DIM;

  // Issue the first buffer's four tile loads (Xh, Xl, Yh, Yl) from wave 0.
  if (wave == 0) {
    tdm_load_tile(smem, 0 * TILE_B, xh + xoff);
    tdm_load_tile(smem, 1 * TILE_B, xl + xoff);
    tdm_load_tile(smem, 2 * TILE_B, yh + yoff);
    tdm_load_tile(smem, 3 * TILE_B, yl + yoff);
    __builtin_amdgcn_s_wait_tensorcnt(0);
  }
  __syncthreads();

  v8f acc[2][4] = {};
  int cur = 0;

  for (int kk = 0; kk < K_DIM; kk += KTILE) {
    const bool more = (kk + KTILE) < K_DIM;
    // Prefetch next k-slice into the other buffer while we compute.
    if (wave == 0 && more) {
      const unsigned nb = (unsigned)((cur ^ 1) * BUF_B);
      tdm_load_tile(smem, nb + 0 * TILE_B, xh + xoff + kk + KTILE);
      tdm_load_tile(smem, nb + 1 * TILE_B, xl + xoff + kk + KTILE);
      tdm_load_tile(smem, nb + 2 * TILE_B, yh + yoff + kk + KTILE);
      tdm_load_tile(smem, nb + 3 * TILE_B, yl + yoff + kk + KTILE);
    }

    const __bf16* Xh = smem + (size_t)cur * (BUF_B / 2);
    const __bf16* Xl = Xh + TILE * LDSK;
    const __bf16* Yh = Xl + TILE * LDSK;
    const __bf16* Yl = Yh + TILE * LDSK;

    // --- A fragments (16-bit A layout: lanes 0-15 = rows M=0-15 holding
    //     K{0-7,16-23}; lanes 16-31 same rows, K{8-15,24-31}) ---
    Frag ah[2], al[2];
    #pragma unroll
    for (int mi = 0; mi < 2; ++mi) {
      const int r = wm * 32 + mi * 16 + laneRow;
      const __bf16* ph = Xh + r * LDSK + half * 8;
      ah[mi].h[0] = *(const v8bf*)(ph);
      ah[mi].h[1] = *(const v8bf*)(ph + 16);
      const __bf16* pl = Xl + r * LDSK + half * 8;
      al[mi].h[0] = *(const v8bf*)(pl);
      al[mi].h[1] = *(const v8bf*)(pl + 16);
    }
    // --- B fragments (B 32x16: lane holds column N=lane&15; lanes 0-15
    //     K{0-7,16-23}, lanes 16-31 K{8-15,24-31}, same as A) ---
    Frag bh[4], bl[4];
    #pragma unroll
    for (int ni = 0; ni < 4; ++ni) {
      const int r = wn * 64 + ni * 16 + laneRow;
      const __bf16* ph = Yh + r * LDSK + half * 8;
      bh[ni].h[0] = *(const v8bf*)(ph);
      bh[ni].h[1] = *(const v8bf*)(ph + 16);
      const __bf16* pl = Yl + r * LDSK + half * 8;
      bl[ni].h[0] = *(const v8bf*)(pl);
      bl[ni].h[1] = *(const v8bf*)(pl + 16);
    }

    // --- bf16x3 split-precision MACs: hh + hl + lh ---
    #pragma unroll
    for (int mi = 0; mi < 2; ++mi) {
      #pragma unroll
      for (int ni = 0; ni < 4; ++ni) {
        acc[mi][ni] = __builtin_amdgcn_wmma_f32_16x16x32_bf16(
            false, ah[mi].v, false, bh[ni].v, (short)0, acc[mi][ni], false, false);
        acc[mi][ni] = __builtin_amdgcn_wmma_f32_16x16x32_bf16(
            false, ah[mi].v, false, bl[ni].v, (short)0, acc[mi][ni], false, false);
        acc[mi][ni] = __builtin_amdgcn_wmma_f32_16x16x32_bf16(
            false, al[mi].v, false, bh[ni].v, (short)0, acc[mi][ni], false, false);
      }
    }

    // Next buffer must be resident before anyone reads it next iteration.
    if (wave == 0 && more) __builtin_amdgcn_s_wait_tensorcnt(0);
    __syncthreads();
    cur ^= 1;
  }

  // --- fused epilogue: out = exp(-gamma * max(xs + ys - 2 s, 0)) ---
  // C layout: VGPR r, lanes 0-15 -> (M=r, N=lane); lanes 16-31 -> (M=r+8, N=lane-16)
  #pragma unroll
  for (int mi = 0; mi < 2; ++mi) {
    const int gm0 = bm + wm * 32 + mi * 16 + half * 8;
    #pragma unroll
    for (int ni = 0; ni < 4; ++ni) {
      const int gn = bn + wn * 64 + ni * 16 + laneRow;
      const float ys = ysq[gn];
      const v8f c = acc[mi][ni];
      #pragma unroll
      for (int r = 0; r < 8; ++r) {
        float d = xsq[gm0 + r] + ys - 2.0f * c[r];
        d = fmaxf(d, 0.0f);
        __builtin_nontemporal_store(__expf(-GAMMA * d),
                                    &out[(size_t)(gm0 + r) * ldo + gn]);
      }
    }
  }
}

// ---------------------------------------------------------------------------
extern "C" void kernel_launch(void* const* d_in, const int* in_sizes, int n_in,
                              void* d_out, int out_size, void* d_ws, size_t ws_size,
                              hipStream_t stream) {
  const float* x = (const float*)d_in[0];
  const float* y = (const float*)d_in[1];
  float* out = (float*)d_out;

  const int nx = in_sizes[0] / K_DIM;   // 8192
  const int ny = in_sizes[1] / K_DIM;   // 8192
  const size_t elems = (size_t)N_ROWS * K_DIM;

  // Workspace layout: norms (64 KB) then 4 bf16 matrices (32 MB total).
  float*  xsq = (float*)d_ws;           // [nx]
  float*  ysq = xsq + nx;               // [ny]
  __bf16* Xh  = (__bf16*)(ysq + ny);
  __bf16* Xl  = Xh + elems;
  __bf16* Yh  = Xl + elems;
  __bf16* Yl  = Yh + elems;

  const int cblocks = (int)(elems / (256 * 8));
  split_bf16_kernel<<<cblocks, 256, 0, stream>>>(x, Xh, Xl);
  split_bf16_kernel<<<cblocks, 256, 0, stream>>>(y, Yh, Yl);
  row_sqnorm_kernel<<<nx / 8, 256, 0, stream>>>(x, xsq);
  row_sqnorm_kernel<<<ny / 8, 256, 0, stream>>>(y, ysq);

  dim3 grid(nx / TILE, ny / TILE);      // 64 x 64 workgroups
  rbf_wmma_kernel<<<grid, 256, 0, stream>>>(Xh, Xl, Yh, Yl, xsq, ysq, out, ny);
}
